// GATNoEdgeAttrs_89240830476321
// MI455X (gfx1250) — matrix operand order repaired and checked by reading, not compiled, not run on hardware
//
#include <hip/hip_runtime.h>
#include <math.h>

// ---------------- types ----------------
typedef __attribute__((ext_vector_type(16))) __bf16 v16bf;
typedef __attribute__((ext_vector_type(8)))  __bf16 v8bf;
typedef __attribute__((ext_vector_type(4)))  __bf16 v4bf;
typedef __attribute__((ext_vector_type(8)))  float  v8f;

#define NNODES 100000
#define NEDGES 1600000
#define NHEADS 4
#define NB     64
#define NOUT   32

__device__ __forceinline__ __bf16 f2bf(float f) {
  unsigned u = __float_as_uint(f);
  unsigned r = (u + 0x7FFFu + ((u >> 16) & 1u)) >> 16;   // round-to-nearest-even
  unsigned short h = (unsigned short)r;
  return __builtin_bit_cast(__bf16, h);
}

// order-preserving float <-> uint encoding for atomicMax on floats
__device__ __forceinline__ unsigned encf(float f) {
  unsigned u = __float_as_uint(f);
  return (u & 0x80000000u) ? ~u : (u | 0x80000000u);
}
__device__ __forceinline__ float decf(unsigned u) {
  u = (u & 0x80000000u) ? (u & 0x7FFFFFFFu) : ~u;
  return __uint_as_float(u);
}

__device__ __forceinline__ float lrelu(float x) { return x > 0.f ? x : 0.2f * x; }

// ---------------- WMMA GEMM: O[N,M] = X[N,K] @ W[K,M]  (bf16 compute, f32 accum) ----------------
// Compile-time K,M. 256 threads = 8 waves; wave w owns rows [w*16, w*16+16), up to 4 col tiles.
// LDS: sX row-major [128][KP]; sWt holds W TRANSPOSED [MP][KP] so B fragments are contiguous.
template <int K, int M>
__global__ __launch_bounds__(256) void k_gemm_wmma(const float* __restrict__ X,
                                                   const float* __restrict__ W,
                                                   float* __restrict__ O, int n) {
  constexpr int KP = (K + 31) & ~31;     // 128,64,32,32
  constexpr int MP = (M + 15) & ~15;     // 64,32,16,16
  constexpr int colTiles = MP / 16;
  constexpr int kchunks  = KP / 32;
  constexpr int KV = K / 4;              // float4 chunks per row

  __shared__ __bf16 sX[128 * KP];
  __shared__ __bf16 sWt[MP * KP];

  const int tid     = threadIdx.x;
  const int rowBase = blockIdx.x * 128;

  // --- X tile -> LDS via float4 global loads, f32->bf16 packed 4-wide ---
  for (int idx = tid; idx < 128 * KV; idx += 256) {
    int r = idx / KV, kq = (idx % KV) * 4;
    int gr = rowBase + r;
    float4 v = make_float4(0.f, 0.f, 0.f, 0.f);
    if (gr < n) v = *(const float4*)(X + (size_t)gr * K + kq);
    v4bf p;
    p[0] = f2bf(v.x); p[1] = f2bf(v.y); p[2] = f2bf(v.z); p[3] = f2bf(v.w);
    *(v4bf*)(&sX[r * KP + kq]) = p;
  }
  if constexpr (KP > K) {                // zero-pad K range (layer 4)
    for (int idx = tid; idx < 128 * (KP - K); idx += 256) {
      int r = idx / (KP - K), k = K + idx % (KP - K);
      sX[r * KP + k] = f2bf(0.f);
    }
  }
  // --- W -> LDS transposed [c][k], zero-padded ---
  for (int idx = tid; idx < MP * KP; idx += 256) {
    int c = idx / KP, k = idx % KP;
    float v = (k < K && c < M) ? W[k * M + c] : 0.f;
    sWt[c * KP + k] = f2bf(v);
  }
  __syncthreads();

  const int wv = tid >> 5, lane = tid & 31;
  const int half = lane >> 4, m16 = lane & 15;

  v8f acc[colTiles];
#pragma unroll
  for (int ct = 0; ct < colTiles; ++ct)
#pragma unroll
    for (int r = 0; r < 8; ++r) acc[ct][r] = 0.f;

  const __bf16* aRow = &sX[(wv * 16 + m16) * KP];

#pragma unroll
  for (int kc = 0; kc < kchunks; ++kc) {
    const int kb = kc * 32;
    // A fragment (16x32 bf16 ISA layout): two contiguous 8-element runs per lane
    //   elems 0..7  -> K = kb + half*8 + {0..7}
    //   elems 8..15 -> K = kb + 16 + half*8 + {0..7}
    v8bf a0 = *(const v8bf*)(aRow + kb + half * 8);
    v8bf a1 = *(const v8bf*)(aRow + kb + 16 + half * 8);
    v16bf a = __builtin_shufflevector(a0, a1, 0, 1, 2, 3, 4, 5, 6, 7,
                                      8, 9, 10, 11, 12, 13, 14, 15);
#pragma unroll
    for (int ct = 0; ct < colTiles; ++ct) {
      // B fragment (32x16): lane holds col n=ct*16+m16, elems 0..7 -> K = kb + half*8 + {0..7}, elems 8..15 -> K = kb + 16 + half*8 + {0..7}
      const __bf16* bRow = &sWt[(ct * 16 + m16) * KP + kb + half * 8];
      v8bf b0 = *(const v8bf*)(bRow);
      v8bf b1 = *(const v8bf*)(bRow + 16);
      v16bf b = __builtin_shufflevector(b0, b1, 0, 1, 2, 3, 4, 5, 6, 7,
                                        8, 9, 10, 11, 12, 13, 14, 15);
      acc[ct] = __builtin_amdgcn_wmma_f32_16x16x32_bf16(
          /*neg_a=*/false, a, /*neg_b=*/false, b,
          /*c_mod=*/(short)0, acc[ct], /*reuse_a=*/false, /*reuse_b=*/false);
    }
  }

  // C/D layout: VGPR r -> row (r + 8*half), col m16
#pragma unroll
  for (int ct = 0; ct < colTiles; ++ct) {
#pragma unroll
    for (int r = 0; r < 8; ++r) {
      int gr = rowBase + wv * 16 + r + 8 * half;
      int gc = ct * 16 + m16;
      if (gr < n && gc < M) O[(size_t)gr * M + gc] = acc[ct][r];
    }
  }
}

// ---------------- attention logits: al_s[n,h] = <H[n,h,:], a_src[h,:]> ----------------
__global__ void k_att(const float* __restrict__ g, const float* __restrict__ a_src,
                      const float* __restrict__ a_dst, float* __restrict__ al_s,
                      float* __restrict__ al_d, int n, int oc) {
  int i = blockIdx.x * blockDim.x + threadIdx.x;
  if (i >= n * NHEADS) return;
  int nd = i >> 2, h = i & 3;
  const float* row = g + (size_t)nd * (NHEADS * oc) + h * oc;
  float ss = 0.f, dd = 0.f;
  for (int c = 0; c < oc; ++c) {
    float v = row[c];
    ss += v * a_src[h * oc + c];
    dd += v * a_dst[h * oc + c];
  }
  al_s[i] = ss;
  al_d[i] = dd;
}

// ---------------- edge pass 1: per-destination running max (ordered-uint atomicMax) ----------------
__global__ void k_edge_max(const int* __restrict__ ei, const float* __restrict__ al_s,
                           const float* __restrict__ al_d, unsigned* __restrict__ menc) {
  int i = blockIdx.x * blockDim.x + threadIdx.x;
  if (i >= NEDGES + NNODES) return;
  int s, d;
  if (i < NEDGES) { s = ei[i]; d = ei[NEDGES + i]; }
  else            { s = d = i - NEDGES; }            // self-loops
#pragma unroll
  for (int h = 0; h < NHEADS; ++h) {
    float e = lrelu(al_s[s * NHEADS + h] + al_d[d * NHEADS + h]);
    atomicMax(&menc[d * NHEADS + h], encf(e));
  }
}

// ---------------- edge pass 2: ex = exp(e - m); s += ex; acc[dst] += ex * H[src] ----------------
__global__ void k_edge_acc(const int* __restrict__ ei, const float* __restrict__ al_s,
                           const float* __restrict__ al_d, const unsigned* __restrict__ menc,
                           float* __restrict__ ssum, const float* __restrict__ g,
                           float* __restrict__ acc, int oc) {
  int t = blockIdx.x * blockDim.x + threadIdx.x;
  if (t >= (NEDGES + NNODES) * NHEADS) return;
  int i = t >> 2, h = t & 3;
  int s, d;
  if (i < NEDGES) { s = ei[i]; d = ei[NEDGES + i]; }
  else            { s = d = i - NEDGES; }
  float e  = lrelu(al_s[s * NHEADS + h] + al_d[d * NHEADS + h]);
  float mx = decf(menc[d * NHEADS + h]);
  float ex = __expf(e - mx);
  atomicAdd(&ssum[d * NHEADS + h], ex);
  const int HOC = NHEADS * oc;
  const float* grow = g + (size_t)s * HOC + h * oc;
  float* arow = acc + (size_t)d * HOC + h * oc;
  for (int c = 0; c < oc; ++c) atomicAdd(&arow[c], ex * grow[c]);
}

// ---------------- finalize: out = relu(acc / (s + 1e-16) + bias), in place ----------------
__global__ void k_fin(float* __restrict__ acc, const float* __restrict__ ssum,
                      const float* __restrict__ bias, int n, int oc) {
  int HOC = NHEADS * oc;
  int i = blockIdx.x * blockDim.x + threadIdx.x;
  if (i >= n * HOC) return;
  int nd = i / HOC, j = i % HOC, h = j / oc;
  float v = acc[i] / (ssum[nd * NHEADS + h] + 1e-16f) + bias[j];
  acc[i] = v > 0.f ? v : 0.f;
}

// ---------------- global mean pool (segment sum + count) ----------------
__global__ void k_pool(const float* __restrict__ h, const int* __restrict__ batch,
                       float* __restrict__ pooled, float* __restrict__ cnt) {
  int i = blockIdx.x * blockDim.x + threadIdx.x;
  if (i >= NNODES) return;
  int b = batch[i];
#pragma unroll
  for (int c = 0; c < 8; ++c) atomicAdd(&pooled[b * 8 + c], h[(size_t)i * 8 + c]);
  atomicAdd(&cnt[b], 1.f);
}

// ---------------- final FC: out[b,o] = (pooled[b,:]/max(cnt,1)) @ Wfc + bfc ----------------
__global__ void k_fc(const float* __restrict__ pooled, const float* __restrict__ cnt,
                     const float* __restrict__ Wfc, const float* __restrict__ bfc,
                     float* __restrict__ out) {
  int t = blockIdx.x * blockDim.x + threadIdx.x;
  if (t >= NB * NOUT) return;
  int b = t >> 5, o = t & 31;
  float cv = cnt[b];
  cv = cv > 1.f ? cv : 1.f;
  float s = bfc[o];
#pragma unroll
  for (int c = 0; c < 8; ++c) s += (pooled[b * 8 + c] / cv) * Wfc[c * NOUT + o];
  out[t] = s;
}

// ---------------- host-side launcher ----------------
extern "C" void kernel_launch(void* const* d_in, const int* in_sizes, int n_in,
                              void* d_out, int out_size, void* d_ws, size_t ws_size,
                              hipStream_t stream) {
  (void)in_sizes; (void)n_in; (void)out_size; (void)ws_size;

  const float* x     = (const float*)d_in[0];
  const int*   ei    = (const int*)d_in[1];   // [2, E]
  const int*   batch = (const int*)d_in[2];
  const float* Wfc   = (const float*)d_in[19];
  const float* bfc   = (const float*)d_in[20];
  float*       out   = (float*)d_out;

  // workspace carve-up
  char* ws = (char*)d_ws;
  size_t off = 0;
  float*    h0     = (float*)(ws + off); off += (size_t)NNODES * 64 * 4;  // GEMM output H
  float*    h1     = (float*)(ws + off); off += (size_t)NNODES * 64 * 4;  // message accum / layer result
  float*    al_s   = (float*)(ws + off); off += (size_t)NNODES * NHEADS * 4;
  float*    al_d   = (float*)(ws + off); off += (size_t)NNODES * NHEADS * 4;
  unsigned* menc   = (unsigned*)(ws + off); off += (size_t)NNODES * NHEADS * 4;
  float*    ssum   = (float*)(ws + off); off += (size_t)NNODES * NHEADS * 4;
  float*    pooled = (float*)(ws + off); off += (size_t)NB * 8 * 4;
  float*    cnt    = (float*)(ws + off); off += (size_t)NB * 4;

  const int OCs[4] = {16, 8, 4, 2};

  const int gemmBlocks = (NNODES + 127) / 128;
  const int attBlocks  = (NNODES * NHEADS + 255) / 256;
  const int emBlocks   = (NEDGES + NNODES + 255) / 256;
  const int eaBlocks   = ((NEDGES + NNODES) * NHEADS + 255) / 256;

  const float* in = x;
  for (int l = 0; l < 4; ++l) {
    const int oc = OCs[l], M = NHEADS * oc;
    const float* W   = (const float*)d_in[3 + 4 * l];
    const float* a_s = (const float*)d_in[4 + 4 * l];
    const float* a_d = (const float*)d_in[5 + 4 * l];
    const float* b   = (const float*)d_in[6 + 4 * l];

    switch (l) {
      case 0: k_gemm_wmma<128, 64><<<gemmBlocks, 256, 0, stream>>>(in, W, h0, NNODES); break;
      case 1: k_gemm_wmma< 64, 32><<<gemmBlocks, 256, 0, stream>>>(in, W, h0, NNODES); break;
      case 2: k_gemm_wmma< 32, 16><<<gemmBlocks, 256, 0, stream>>>(in, W, h0, NNODES); break;
      default: k_gemm_wmma<16, 8><<<gemmBlocks, 256, 0, stream>>>(in, W, h0, NNODES); break;
    }
    // input buffer (h1) is dead after the GEMM above -> reuse as accumulator
    hipMemsetAsync(h1, 0, (size_t)NNODES * 64 * 4, stream);
    hipMemsetAsync(menc, 0, (size_t)NNODES * NHEADS * 4, stream);  // 0u == encoded minimum
    hipMemsetAsync(ssum, 0, (size_t)NNODES * NHEADS * 4, stream);

    k_att<<<attBlocks, 256, 0, stream>>>(h0, a_s, a_d, al_s, al_d, NNODES, oc);
    k_edge_max<<<emBlocks, 256, 0, stream>>>(ei, al_s, al_d, menc);
    k_edge_acc<<<eaBlocks, 256, 0, stream>>>(ei, al_s, al_d, menc, ssum, h0, h1, oc);

    const int finBlocks = (NNODES * M + 255) / 256;
    k_fin<<<finBlocks, 256, 0, stream>>>(h1, ssum, b, NNODES, oc);
    in = h1;
  }

  hipMemsetAsync(pooled, 0, (size_t)NB * 8 * 4, stream);
  hipMemsetAsync(cnt, 0, (size_t)NB * 4, stream);
  k_pool<<<(NNODES + 255) / 256, 256, 0, stream>>>(h1, batch, pooled, cnt);
  k_fc<<<(NB * NOUT + 255) / 256, 256, 0, stream>>>(pooled, cnt, Wfc, bfc, out);
}
